// LigandEncoder_32263794327823
// MI455X (gfx1250) — matrix operand order repaired and checked by reading, not compiled, not run on hardware
//
#include <hip/hip_runtime.h>
#include <hip/hip_bf16.h>

typedef __bf16 bf16;
typedef __attribute__((ext_vector_type(8)))  bf16  v8bf;
typedef __attribute__((ext_vector_type(16))) bf16  v16bf;
typedef __attribute__((ext_vector_type(8)))  float v8f;

#define F_DIM 128   // per-head feature width (HID == OUT == 128)

__device__ __forceinline__ float lrelu(float x) { return x > 0.f ? x : 0.2f * x; }

// Edge e in [0, E) comes from edge_index; e in [E, E+N) is self-loop (e-E, e-E).
__device__ __forceinline__ void edge_sd(const int* ei, int E, int e, int& s, int& d) {
  if (e < E) { s = ei[e]; d = ei[E + e]; } else { s = e - E; d = s; }
}

// IEEE-ordered float atomic max via int/uint reinterpretation.
__device__ __forceinline__ void atomicMaxF(float* addr, float val) {
  if (val >= 0.f) atomicMax((int*)addr, __float_as_int(val));
  else            atomicMin((unsigned int*)addr, __float_as_uint(val));
}

// Load a 16-element bf16 fragment as two 16B chunks (lo at p, hi at p+hiOff).
__device__ __forceinline__ v16bf load_frag(const bf16* p, int hiOff) {
  v8bf lo = *(const v8bf*)(p);
  v8bf hi = *(const v8bf*)(p + hiOff);
  return __builtin_shufflevector(lo, hi, 0,1,2,3,4,5,6,7,8,9,10,11,12,13,14,15);
}

// ---------------- GEMM: C[M,Nout](bf16) = A[M,K](bf16) @ Wt[Nout,K]^T, fp32 accum ----
// Block = 128 threads = 4 waves. Block tile: 32 rows x 128 cols; wave tile 32x32.
__global__ void k_gemm_bf16(const bf16* __restrict__ A, const bf16* __restrict__ Wt,
                            bf16* __restrict__ C, int M, int K, int Nout) {
  const int lane = threadIdx.x & 31;
  const int wave = threadIdx.x >> 5;
  const int half = lane >> 4;   // which 16-lane half
  const int lid  = lane & 15;
  const int mBase = blockIdx.x * 32;
  const int nBase = blockIdx.y * 128 + wave * 32;

  int rA0 = mBase + lid;       if (rA0 > M - 1) rA0 = M - 1;
  int rA1 = mBase + 16 + lid;  if (rA1 > M - 1) rA1 = M - 1;
  const bf16* pa0 = A  + (size_t)rA0 * K;
  const bf16* pa1 = A  + (size_t)rA1 * K;
  const bf16* pb0 = Wt + (size_t)(nBase + lid) * K;        // column n = nBase+lid
  const bf16* pb1 = Wt + (size_t)(nBase + 16 + lid) * K;

  v8f acc00 = {}, acc01 = {}, acc10 = {}, acc11 = {};
  for (int k0 = 0; k0 < K; k0 += 32) {
    // A 16x32: elems 0..7 -> K = k0 + half*8 + j ; elems 8..15 -> +16
    const int koA = k0 + half * 8;
    v16bf a0 = load_frag(pa0 + koA, 16);
    v16bf a1 = load_frag(pa1 + koA, 16);
    // B 32x16: elems 0..7 -> K = k0 + half*8 + j ; elems 8..15 -> +16 (same as A)
    const int koB = k0 + half * 8;
    v16bf b0 = load_frag(pb0 + koB, 16);
    v16bf b1 = load_frag(pb1 + koB, 16);
    acc00 = __builtin_amdgcn_wmma_f32_16x16x32_bf16(false, a0, false, b0, (short)0, acc00, false, false);
    acc01 = __builtin_amdgcn_wmma_f32_16x16x32_bf16(false, a0, false, b1, (short)0, acc01, false, false);
    acc10 = __builtin_amdgcn_wmma_f32_16x16x32_bf16(false, a1, false, b0, (short)0, acc10, false, false);
    acc11 = __builtin_amdgcn_wmma_f32_16x16x32_bf16(false, a1, false, b1, (short)0, acc11, false, false);
  }
  const int col0 = nBase + lid, col1 = col0 + 16;
  #pragma unroll
  for (int r = 0; r < 8; ++r) {
    int row0 = mBase + half * 8 + r;
    int row1 = row0 + 16;
    if (row0 < M) {
      C[(size_t)row0 * Nout + col0] = (bf16)acc00[r];
      C[(size_t)row0 * Nout + col1] = (bf16)acc01[r];
    }
    if (row1 < M) {
      C[(size_t)row1 * Nout + col0] = (bf16)acc10[r];
      C[(size_t)row1 * Nout + col1] = (bf16)acc11[r];
    }
  }
}

// ---------------- small utility kernels ----------------
__global__ void k_f2bf(const float* __restrict__ x, bf16* __restrict__ o, int n) {
  int t = blockIdx.x * blockDim.x + threadIdx.x;
  if (t < n) o[t] = (bf16)x[t];
}
__global__ void k_zero_f(float* __restrict__ o, int n) {
  int t = blockIdx.x * blockDim.x + threadIdx.x;
  if (t < n) o[t] = 0.f;
}
// Transpose weight W[K,Nout] (fp32) -> Wt[Nout,K] (bf16)
__global__ void k_wt(const float* __restrict__ W, bf16* __restrict__ Wt, int K, int Nout) {
  int t = blockIdx.x * blockDim.x + threadIdx.x;
  if (t >= K * Nout) return;
  int n = t / K, k = t - n * K;
  Wt[(size_t)n * K + k] = (bf16)W[(size_t)k * Nout + n];
}
__global__ void k_init_md(float* __restrict__ m, float* __restrict__ den, int n) {
  int t = blockIdx.x * blockDim.x + threadIdx.x;
  if (t < n) { m[t] = -1e30f; den[t] = 0.f; }
}

// al_s[n,h] = dot(h[n,h,:], a_s[h,:]) ; al_d likewise
__global__ void k_logits(const bf16* __restrict__ Hb, const float* __restrict__ as_,
                         const float* __restrict__ ad_, float* __restrict__ als,
                         float* __restrict__ ald, int Nn, int H, int Kout) {
  int t = blockIdx.x * blockDim.x + threadIdx.x;
  if (t >= Nn * H) return;
  int n = t / H, h = t - n * H;
  const bf16*  hr = Hb  + (size_t)n * Kout + h * F_DIM;
  const float* vs = as_ + h * F_DIM;
  const float* vd = ad_ + h * F_DIM;
  float ss = 0.f, sd = 0.f;
  for (int f = 0; f < F_DIM; ++f) { float v = (float)hr[f]; ss += v * vs[f]; sd += v * vd[f]; }
  als[t] = ss; ald[t] = sd;
}

// pass 1: logit = leaky(al_s[src]+al_d[dst]); store; segment max into m[dst]
__global__ void k_edge_logit(const int* __restrict__ ei, int E, int total,
                             const float* __restrict__ als, const float* __restrict__ ald,
                             float* __restrict__ m, float* __restrict__ exb, int H) {
  int t = blockIdx.x * blockDim.x + threadIdx.x;
  if (t >= total) return;
  int e = t / H, h = t - e * H;
  int s, d; edge_sd(ei, E, e, s, d);
  float v = lrelu(als[(size_t)s * H + h] + ald[(size_t)d * H + h]);
  exb[t] = v;
  atomicMaxF(&m[(size_t)d * H + h], v);
}
// pass 2: ex = exp(logit - m[dst]); den[dst] += ex
__global__ void k_edge_exp(const int* __restrict__ ei, int E, int total,
                           const float* __restrict__ m, float* __restrict__ den,
                           float* __restrict__ exb, int H) {
  int t = blockIdx.x * blockDim.x + threadIdx.x;
  if (t >= total) return;
  int e = t / H, h = t - e * H;
  int s, d; edge_sd(ei, E, e, s, d);
  float ev = __expf(exb[t] - m[(size_t)d * H + h]);
  exb[t] = ev;
  atomicAdd(&den[(size_t)d * H + h], ev);
}

// one wave per edge: agg[dst, f] += h[src, f] * alpha[head(f)]
__global__ void k_message(const bf16* __restrict__ Hb, const int* __restrict__ ei, int E, int Etot,
                          const float* __restrict__ exb, const float* __restrict__ den,
                          float* __restrict__ agg, int H, int Kout) {
  int gw = (int)((blockIdx.x * blockDim.x + threadIdx.x) >> 5);
  int lane = threadIdx.x & 31;
  if (gw >= Etot) return;
  int s, d; edge_sd(ei, E, gw, s, d);
  const bf16* hs = Hb + (size_t)s * Kout;
  float* ag = agg + (size_t)d * Kout;
  const int iters = Kout >> 6;                  // 64 features per wave step (2 per lane)
  for (int i = 0; i < iters; ++i) {
    int f = i * 64 + lane * 2;
    int h = f >> 7;                             // F_DIM == 128
    float alpha = exb[(size_t)gw * H + h] / (den[(size_t)d * H + h] + 1e-16f);
    atomicAdd(&ag[f],     (float)hs[f]     * alpha);
    atomicAdd(&ag[f + 1], (float)hs[f + 1] * alpha);
  }
}

// layers 1,2: next input = bf16(leaky(agg + b))
__global__ void k_finalize(const float* __restrict__ agg, const float* __restrict__ b,
                           bf16* __restrict__ inA, int n, int Kout) {
  int t = blockIdx.x * blockDim.x + threadIdx.x;
  if (t >= n) return;
  int c = t % Kout;
  inA[t] = (bf16)lrelu(agg[t] + b[c]);
}

// pooling helpers
__global__ void k_count(const int* __restrict__ batch, float* __restrict__ gcnt, int Nn) {
  int t = blockIdx.x * blockDim.x + threadIdx.x;
  if (t < Nn) atomicAdd(&gcnt[batch[t]], 1.0f);
}
// layer 3: val = leaky(agg + b3); gsum[batch[n]] += val
__global__ void k_final_pool(const float* __restrict__ agg, const float* __restrict__ b,
                             const int* __restrict__ batch, float* __restrict__ gsum, int Nn) {
  int t = blockIdx.x * blockDim.x + threadIdx.x;
  if (t >= Nn * F_DIM) return;
  int n = t >> 7, c = t & 127;
  float val = lrelu(agg[t] + b[c]);
  atomicAdd(&gsum[(size_t)batch[n] * F_DIM + c], val);
}
__global__ void k_pool_div(const float* __restrict__ gsum, const float* __restrict__ gcnt,
                           float* __restrict__ out, int G) {
  int t = blockIdx.x * blockDim.x + threadIdx.x;
  if (t >= G * F_DIM) return;
  out[t] = gsum[t] / fmaxf(gcnt[t >> 7], 1.0f);
}

extern "C" void kernel_launch(void* const* d_in, const int* in_sizes, int n_in,
                              void* d_out, int out_size, void* d_ws, size_t ws_size,
                              hipStream_t stream) {
  (void)n_in; (void)ws_size;
  const float* x     = (const float*)d_in[0];
  const int*   ei    = (const int*)d_in[1];
  const int*   batch = (const int*)d_in[2];
  const float* W[3]  = {(const float*)d_in[3], (const float*)d_in[7],  (const float*)d_in[11]};
  const float* As[3] = {(const float*)d_in[4], (const float*)d_in[8],  (const float*)d_in[12]};
  const float* Ad[3] = {(const float*)d_in[5], (const float*)d_in[9],  (const float*)d_in[13]};
  const float* Bi[3] = {(const float*)d_in[6], (const float*)d_in[10], (const float*)d_in[14]};

  const int Nn   = in_sizes[2];          // 200000 nodes
  const int E    = in_sizes[1] / 2;      // 1.6M directed edges
  const int Etot = E + Nn;               // + self loops
  const int G    = out_size / F_DIM;     // 5000 graphs
  const int KinArr[3]  = {in_sizes[0] / Nn, in_sizes[6],  in_sizes[10]}; // 128, 384, 384
  const int KoutArr[3] = {in_sizes[6],      in_sizes[10], in_sizes[14]}; // 384, 384, 128
  const int Harr[3]    = {KoutArr[0] / F_DIM, KoutArr[1] / F_DIM, KoutArr[2] / F_DIM}; // 3,3,1

  // ---- carve workspace (all buffers fully initialized below) ----
  char* p = (char*)d_ws;
  auto carve = [&](size_t bytes) { char* r = p; p += (bytes + 255) & ~(size_t)255; return (void*)r; };
  bf16*  inA  = (bf16*)carve((size_t)Nn * 384 * sizeof(bf16));   // layer input (bf16)
  bf16*  Hb   = (bf16*)carve((size_t)Nn * 384 * sizeof(bf16));   // h = X@W (bf16, L2-resident)
  float* agg  = (float*)carve((size_t)Nn * 384 * sizeof(float)); // fp32 scatter accumulator
  float* als  = (float*)carve((size_t)Nn * 3 * sizeof(float));
  float* ald  = (float*)carve((size_t)Nn * 3 * sizeof(float));
  float* mbuf = (float*)carve((size_t)Nn * 3 * sizeof(float));
  float* den  = (float*)carve((size_t)Nn * 3 * sizeof(float));
  float* exb  = (float*)carve((size_t)Etot * 3 * sizeof(float)); // edge logits, then exp
  bf16*  Wt   = (bf16*)carve((size_t)384 * 384 * sizeof(bf16));  // transposed bf16 weights
  float* gsum = (float*)carve((size_t)G * F_DIM * sizeof(float));
  float* gcnt = (float*)carve((size_t)G * sizeof(float));

  auto nb = [](long long n, int b) { return (unsigned)((n + b - 1) / b); };

  // input features -> bf16
  k_f2bf<<<nb((long long)Nn * KinArr[0], 256), 256, 0, stream>>>(x, inA, Nn * KinArr[0]);
  // graph counts (once)
  k_zero_f<<<nb((long long)G * F_DIM, 256), 256, 0, stream>>>(gsum, G * F_DIM);
  k_zero_f<<<nb(G, 256), 256, 0, stream>>>(gcnt, G);
  k_count<<<nb(Nn, 256), 256, 0, stream>>>(batch, gcnt, Nn);

  for (int l = 0; l < 3; ++l) {
    const int K = KinArr[l], Ko = KoutArr[l], H = Harr[l];
    k_wt<<<nb((long long)K * Ko, 256), 256, 0, stream>>>(W[l], Wt, K, Ko);
    dim3 gg(nb(Nn, 32), Ko / 128);
    k_gemm_bf16<<<gg, 128, 0, stream>>>(inA, Wt, Hb, Nn, K, Ko);
    k_zero_f<<<nb((long long)Nn * Ko, 256), 256, 0, stream>>>(agg, Nn * Ko);
    k_init_md<<<nb((long long)Nn * H, 256), 256, 0, stream>>>(mbuf, den, Nn * H);
    k_logits<<<nb((long long)Nn * H, 256), 256, 0, stream>>>(Hb, As[l], Ad[l], als, ald, Nn, H, Ko);
    k_edge_logit<<<nb((long long)Etot * H, 256), 256, 0, stream>>>(ei, E, Etot * H, als, ald, mbuf, exb, H);
    k_edge_exp<<<nb((long long)Etot * H, 256), 256, 0, stream>>>(ei, E, Etot * H, mbuf, den, exb, H);
    k_message<<<nb((long long)Etot * 32, 256), 256, 0, stream>>>(Hb, ei, E, Etot, exb, den, agg, H, Ko);
    if (l < 2) {
      k_finalize<<<nb((long long)Nn * Ko, 256), 256, 0, stream>>>(agg, Bi[l], inA, Nn * Ko, Ko);
    } else {
      k_final_pool<<<nb((long long)Nn * F_DIM, 256), 256, 0, stream>>>(agg, Bi[l], batch, gsum, Nn);
    }
  }
  k_pool_div<<<nb((long long)G * F_DIM, 256), 256, 0, stream>>>(gsum, gcnt, (float*)d_out, G);
}
